// SubMConv2dInsDilateASPP_25683904430830
// MI455X (gfx1250) — matrix operand order repaired and checked by reading, not compiled, not run on hardware
//
#include <hip/hip_runtime.h>
#include <climits>
#include <cstdint>
#include <cstddef>

// Problem constants (match reference)
#define BN 4
#define HH 512
#define WW 512
#define NPTS 262144
#define CCOUT 64
#define EPSV 1e-5f

typedef __attribute__((ext_vector_type(16))) __bf16 v16bf;
typedef __attribute__((ext_vector_type(8)))  float v8f;

union ABReg { v16bf v; uint4 q[2]; };

// ---------------------------------------------------------------------------
// Small setup kernels
// ---------------------------------------------------------------------------

// bbox layout: [0..15]=ymin [16..31]=ymax [32..47]=xmin [48..63]=xmax [64..79]=cnt
__global__ void init_small(int* bbox) {
    int t = threadIdx.x;
    if (t < 16) {
        bbox[t]      = INT_MAX;
        bbox[16 + t] = INT_MIN;
        bbox[32 + t] = INT_MAX;
        bbox[48 + t] = INT_MIN;
        bbox[64 + t] = 0;
    }
}

__global__ void scatter_idx(const int* __restrict__ indices, int* __restrict__ idx_map) {
    int i = blockIdx.x * 256 + threadIdx.x;
    int b = indices[3 * i + 0];
    int y = indices[3 * i + 1];
    int x = indices[3 * i + 2];
    idx_map[(b * HH + y) * WW + x] = i;
}

__global__ void bbox_kernel(const int* __restrict__ indices, const int* __restrict__ seg,
                            int* __restrict__ bbox) {
    __shared__ int ly[16], lY[16], lx[16], lX[16], lc[16];
    int t = threadIdx.x;
    if (t < 16) { ly[t] = INT_MAX; lY[t] = INT_MIN; lx[t] = INT_MAX; lX[t] = INT_MIN; lc[t] = 0; }
    __syncthreads();
    int i = blockIdx.x * 256 + t;
    int s = seg[i];
    int y = indices[3 * i + 1];
    int x = indices[3 * i + 2];
    atomicMin(&ly[s], y); atomicMax(&lY[s], y);
    atomicMin(&lx[s], x); atomicMax(&lX[s], x);
    atomicAdd(&lc[s], 1);
    __syncthreads();
    if (t < 16 && lc[t] > 0) {
        atomicMin(&bbox[t],      ly[t]);
        atomicMax(&bbox[16 + t], lY[t]);
        atomicMin(&bbox[32 + t], lx[t]);
        atomicMax(&bbox[48 + t], lX[t]);
        atomicAdd(&bbox[64 + t], lc[t]);
    }
}

// dins[r*16+ins] = max(1, ceil(max(hr,wr)*rate)); starts = exclusive prefix of cnt
__global__ void dil_starts(const int* __restrict__ bbox, int* __restrict__ dins,
                           int* __restrict__ starts) {
    int t = threadIdx.x;
    if (t < 48) {
        int ins = t & 15;
        int r   = t >> 4;
        const int rates[3] = {1, 3, 5};
        int cnt = bbox[64 + ins];
        int dv = 1;
        if (cnt > 0) {
            float hr = (float)(bbox[16 + ins] - bbox[ins])      / (float)HH;
            float wr = (float)(bbox[48 + ins] - bbox[32 + ins]) / (float)WW;
            float m  = fmaxf(hr, wr) * (float)rates[r];
            dv = (int)ceilf(m);
            if (dv < 1) dv = 1;
        }
        dins[r * 16 + ins] = dv;
    }
    if (t == 63) {  // seg is sorted -> contiguous instance ranges
        int s = 0;
        for (int j = 0; j < 16; ++j) { starts[j] = s; s += bbox[64 + j]; }
        starts[16] = s;
    }
}

// Pack fp32 weights (9, cin, 64) into WMMA bf16 B-operand panels, laid out for
// bank-conflict-free ds_load_b128:
//   panel p = (tap*kch + chunk)*4 + nt  (512 bf16 = 1024 B each)
//   within panel: offset = q*256 + lane*8 + w   (q = which 16-B beat, w = 0..7)
//   logical: e = q*8+w ; K = chunk*32 + (lane>>4)*16 + e ; N = nt*16 + (lane&15)
__global__ void pack_weights(const float* __restrict__ w, __bf16* __restrict__ out, int kch) {
    int idx = blockIdx.x * 256 + threadIdx.x;
    int total = 9 * kch * 4 * 512;
    if (idx >= total) return;
    int ww_   = idx & 7;
    int lane  = (idx >> 3) & 31;
    int q     = (idx >> 8) & 1;
    int p     = idx >> 9;
    int nt    = p & 3;
    int chunk = (p >> 2) % kch;
    int tap   = (p >> 2) / kch;
    int e = q * 8 + ww_;
    int k = chunk * 32 + (lane >> 4) * 16 + e;
    int n = nt * 16 + (lane & 15);
    int cin = kch * 32;
    out[idx] = (__bf16)w[((size_t)tap * cin + k) * CCOUT + n];
}

// features fp32 (N,64) -> bf16 into cat columns [0,64) of stride 256
__global__ void cvt_feats(const float* __restrict__ f, __bf16* __restrict__ cat) {
    size_t i = (size_t)blockIdx.x * 256 + threadIdx.x;
    size_t pt = i >> 6;
    int c = (int)(i & 63);
    cat[pt * 256 + c] = (__bf16)f[i];
}

// ---------------------------------------------------------------------------
// WMMA submanifold conv: bf16 A/B, f32 accumulate.
// Block = 8 waves; each wave owns 2 M-tiles (32 points) x 64 out channels.
// All B panels staged in LDS once per block (conflict-free layout).
// ---------------------------------------------------------------------------
template <int KCH, bool SAME_INS, bool OUT_F32>
__global__ __launch_bounds__(256) void subm_conv_wmma(
    const __bf16* __restrict__ cat, const int* __restrict__ idx_map,
    const int* __restrict__ indices, const int* __restrict__ seg,
    const int* __restrict__ dins, const __bf16* __restrict__ wpack,
    const float* __restrict__ bias, __bf16* catw, float* outf, int dstOff) {

    extern __shared__ __bf16 lwp[];   // 9*KCH*4 panels * 1024 B

    // Cooperative LDS fill of the whole packed-weight panel set.
    {
        const uint4* src = (const uint4*)wpack;
        uint4* dst = (uint4*)lwp;
        const int total = 9 * KCH * 4 * 64;   // uint4 count (1024 B / panel)
        for (int i = threadIdx.x; i < total; i += 256) dst[i] = src[i];
    }
    __syncthreads();

    const int lane  = threadIdx.x & 31;
    const int wave  = threadIdx.x >> 5;
    const int khalf = lane >> 4;       // which K-half this lane holds (A layout)
    const int mrow  = lane & 15;
    const int tileBase = (blockIdx.x * 8 + wave) * 32;

    int b0[2], yy[2], xx[2], myseg[2], dd[2];
#pragma unroll
    for (int m = 0; m < 2; ++m) {
        const int row = tileBase + m * 16 + mrow;
        b0[m]    = indices[3 * row + 0];
        yy[m]    = indices[3 * row + 1];
        xx[m]    = indices[3 * row + 2];
        myseg[m] = seg[row];
        dd[m]    = SAME_INS ? dins[myseg[m]] : 1;
    }

    v8f acc[2][4];
#pragma unroll
    for (int m = 0; m < 2; ++m)
#pragma unroll
        for (int nt = 0; nt < 4; ++nt) acc[m][nt] = (v8f)0.0f;

    for (int t = 0; t < 9; ++t) {
        const int ky = t / 3 - 1;
        const int kx = t - (t / 3) * 3 - 1;
        int nbr[2];
#pragma unroll
        for (int m = 0; m < 2; ++m) {
            const int ny = yy[m] + ky * dd[m];
            const int nx = xx[m] + kx * dd[m];
            int nb = -1;
            if (ny >= 0 && ny < HH && nx >= 0 && nx < WW)
                nb = idx_map[(b0[m] * HH + ny) * WW + nx];
            if (SAME_INS && nb >= 0 && seg[nb] != myseg[m]) nb = -1;
            nbr[m] = nb;
        }

#pragma unroll
        for (int c = 0; c < KCH; ++c) {
            ABReg a[2];
#pragma unroll
            for (int m = 0; m < 2; ++m) {
                if (nbr[m] >= 0) {
                    // A 16x32 bf16, k order used here (B packed the same): lane<16 holds K {0..15}+kc,
                    // lane>=16 holds K {16..31}+kc (contiguous channel runs)
                    const __bf16* rowp = cat + (size_t)nbr[m] * 256;
                    a[m].q[0] = *(const uint4*)(rowp + c * 32 + 16 * khalf);
                    a[m].q[1] = *(const uint4*)(rowp + c * 32 + 16 * khalf + 8);
                } else {
                    a[m].q[0] = make_uint4(0u, 0u, 0u, 0u);
                    a[m].q[1] = make_uint4(0u, 0u, 0u, 0u);
                }
            }
#pragma unroll
            for (int nt = 0; nt < 4; ++nt) {
                ABReg bb;
                const __bf16* bp = lwp + (size_t)(((t * KCH + c) * 4 + nt) * 512);
                bb.q[0] = *(const uint4*)(bp + lane * 8);          // beat 0
                bb.q[1] = *(const uint4*)(bp + 256 + lane * 8);    // beat 1
                acc[0][nt] = __builtin_amdgcn_wmma_f32_16x16x32_bf16(
                    false, a[0].v, false, bb.v, (short)0, acc[0][nt], false, false);
                acc[1][nt] = __builtin_amdgcn_wmma_f32_16x16x32_bf16(
                    false, a[1].v, false, bb.v, (short)0, acc[1][nt], false, false);
            }
        }
    }

    // C/D layout: lane<16 -> M = v, lane>=16 -> M = v+8 ; N = nt*16 + (lane&15)
#pragma unroll
    for (int m = 0; m < 2; ++m) {
#pragma unroll
        for (int nt = 0; nt < 4; ++nt) {
            const int col = nt * 16 + mrow;
            const float bv = bias[col];
#pragma unroll
            for (int v = 0; v < 8; ++v) {
                const int orow = tileBase + m * 16 + v + 8 * khalf;
                const float val = acc[m][nt][v] + bv;
                if (OUT_F32) outf[(size_t)orow * CCOUT + col] = val;
                else         catw[(size_t)orow * 256 + dstOff + col] = (__bf16)val;
            }
        }
    }
}

// ---------------------------------------------------------------------------
// Deterministic instance-norm statistics: one block per (instance, channel).
// seg sorted -> instance ranges [starts[ins], starts[ins]+cnt[ins]).
// ---------------------------------------------------------------------------
__global__ void stats_kernel(const __bf16* __restrict__ cat, const int* __restrict__ starts,
                             const int* __restrict__ bbox, float* __restrict__ S,
                             float* __restrict__ S2, int dstOff) {
    int ins = blockIdx.x >> 6;
    int c   = blockIdx.x & 63;
    int s0  = starts[ins];
    int n   = bbox[64 + ins];
    float s = 0.f, s2 = 0.f;
    for (int p = threadIdx.x; p < n; p += 256) {
        float v = (float)cat[(size_t)(s0 + p) * 256 + dstOff + c];
        s += v; s2 += v * v;
    }
    __shared__ float rs[256], rq[256];
    rs[threadIdx.x] = s; rq[threadIdx.x] = s2;
    __syncthreads();
    for (int st = 128; st > 0; st >>= 1) {
        if (threadIdx.x < st) {
            rs[threadIdx.x] += rs[threadIdx.x + st];
            rq[threadIdx.x] += rq[threadIdx.x + st];
        }
        __syncthreads();
    }
    if (threadIdx.x == 0) { S[ins * 64 + c] = rs[0]; S2[ins * 64 + c] = rq[0]; }
}

__global__ void norm_kernel(__bf16* cat, const int* __restrict__ seg,
                            const float* __restrict__ S, const float* __restrict__ S2,
                            const int* __restrict__ bbox, int dstOff) {
    size_t idx = (size_t)blockIdx.x * 256 + threadIdx.x;
    size_t pt = idx >> 6;
    int c = (int)(idx & 63);
    int ins = seg[pt];
    float n    = (float)bbox[64 + ins];
    float mean = S[ins * 64 + c] / n;
    float var  = S2[ins * 64 + c] / n - mean * mean;
    float v = (float)cat[pt * 256 + dstOff + c];
    v = (v - mean) * rsqrtf(var + EPSV);
    v = v > 0.f ? v : 0.f;
    cat[pt * 256 + dstOff + c] = (__bf16)v;
}

// ---------------------------------------------------------------------------
// Launch
// ---------------------------------------------------------------------------
extern "C" void kernel_launch(void* const* d_in, const int* in_sizes, int n_in,
                              void* d_out, int out_size, void* d_ws, size_t ws_size,
                              hipStream_t stream) {
    const float* features = (const float*)d_in[0];
    const int*   indices  = (const int*)d_in[1];
    const int*   seg      = (const int*)d_in[2];
    const float* w1 = (const float*)d_in[3];
    const float* b1 = (const float*)d_in[4];
    const float* w2 = (const float*)d_in[5];
    const float* b2 = (const float*)d_in[6];
    const float* w3 = (const float*)d_in[7];
    const float* b3 = (const float*)d_in[8];
    const float* wf = (const float*)d_in[9];
    const float* bf = (const float*)d_in[10];
    float* out = (float*)d_out;

    size_t off = 0;
    auto alloc = [&](size_t bytes) -> char* {
        char* p = (char*)d_ws + off;
        off += (bytes + 255) & ~(size_t)255;
        return p;
    };
    int*    idx_map = (int*)alloc((size_t)BN * HH * WW * 4);
    __bf16* wb0  = (__bf16*)alloc((size_t)9 * 2 * 4 * 512 * 2);
    __bf16* wb1p = (__bf16*)alloc((size_t)9 * 2 * 4 * 512 * 2);
    __bf16* wb2p = (__bf16*)alloc((size_t)9 * 2 * 4 * 512 * 2);
    __bf16* wbf  = (__bf16*)alloc((size_t)9 * 8 * 4 * 512 * 2);
    int*    bbox   = (int*)alloc(5 * 16 * 4);
    int*    dins   = (int*)alloc(48 * 4);
    int*    starts = (int*)alloc(17 * 4);
    float*  S  = (float*)alloc(3 * 1024 * 4);
    float*  S2 = (float*)alloc(3 * 1024 * 4);
    __bf16* cat = (__bf16*)alloc((size_t)NPTS * 256 * 2);

    hipMemsetAsync(idx_map, 0xFF, (size_t)BN * HH * WW * 4, stream);  // -1
    init_small<<<1, 64, 0, stream>>>(bbox);
    scatter_idx<<<NPTS / 256, 256, 0, stream>>>(indices, idx_map);
    bbox_kernel<<<NPTS / 256, 256, 0, stream>>>(indices, seg, bbox);
    dil_starts<<<1, 64, 0, stream>>>(bbox, dins, starts);

    pack_weights<<<(9 * 2 * 4 * 512 + 255) / 256, 256, 0, stream>>>(w1, wb0, 2);
    pack_weights<<<(9 * 2 * 4 * 512 + 255) / 256, 256, 0, stream>>>(w2, wb1p, 2);
    pack_weights<<<(9 * 2 * 4 * 512 + 255) / 256, 256, 0, stream>>>(w3, wb2p, 2);
    pack_weights<<<(9 * 8 * 4 * 512 + 255) / 256, 256, 0, stream>>>(wf, wbf, 8);
    cvt_feats<<<NPTS * 64 / 256, 256, 0, stream>>>(features, cat);

    const size_t ldsBranch = (size_t)9 * 2 * 4 * 512 * 2;  //  73728 B
    const size_t ldsFinal  = (size_t)9 * 8 * 4 * 512 * 2;  // 294912 B

    __bf16* wbs[3] = {wb0, wb1p, wb2p};
    const float* biases[3] = {b1, b2, b3};
    for (int r = 0; r < 3; ++r) {
        subm_conv_wmma<2, true, false><<<NPTS / 256, 256, ldsBranch, stream>>>(
            cat, idx_map, indices, seg, dins + r * 16, wbs[r], biases[r],
            cat, nullptr, 64 + 64 * r);
        stats_kernel<<<1024, 256, 0, stream>>>(cat, starts, bbox, S + r * 1024,
                                               S2 + r * 1024, 64 + 64 * r);
        norm_kernel<<<NPTS * 64 / 256, 256, 0, stream>>>(cat, seg, S + r * 1024,
                                                         S2 + r * 1024, bbox, 64 + 64 * r);
    }

    subm_conv_wmma<8, false, true><<<NPTS / 256, 256, ldsFinal, stream>>>(
        cat, idx_map, indices, seg, dins, wbf, bf, nullptr, out, 0);
}
